// InteractionNet2_49478023250673
// MI455X (gfx1250) — compile-verified, matrix operand order checked by reading, not run on hardware
//
#include <hip/hip_runtime.h>
#include <hip/hip_bf16.h>

// ---------------------------------------------------------------------------
// Fused hypernetwork for MI455X (gfx1250, wave32, WMMA bf16), two passes:
//   pass 1: W (49536x512 f32) -> bf16 in d_ws (memory bound)
//   pass 2: fused  temp = int_x @ W^T + b  +  3-layer per-sample MLP.
// Hyper-weight 16x16 tiles are produced by v_wmma_f32_16x16x32_bf16 straight
// into registers and consumed immediately (temp: 811 MB never materialized).
// Inner loop: dual WMMA accumulation chains, software-pipelined (double-
// buffered) bf16 B loads from L2-resident W and A loads from LDS.
// ---------------------------------------------------------------------------

#define NSAMP   4096
#define KDIM    512                      // INT_DIM
#define DDIM    128                      // hidden width
#define WROWS   (3 * (DDIM + DDIM*DDIM)) // 49536 rows of W
#define MTILE   32                       // samples per workgroup
#define LSTRIDE (DDIM + DDIM * DDIM)     // 16512 temp-columns per layer
#define APAD    520                      // bf16 row stride of int_x tile (bank-conflict free)
#define HPAD    36                       // float row stride of h^T tile  (bank-conflict free)
#define KSTEPS  (KDIM / 32)              // 16 WMMA k-steps

typedef __attribute__((ext_vector_type(16))) __bf16 v16bf;
typedef __attribute__((ext_vector_type(8)))  float  v8f;

union FragB16 {
    v16bf v;
    unsigned u[8];
    uint4 q[2];
};

// round-to-nearest-even f32 -> packed 2x bf16 (one-time paths only)
static __device__ __forceinline__ unsigned pack2_bf16(float a, float b) {
    union { float f; unsigned u; } ua, ub;
    ua.f = a; ub.f = b;
    unsigned ra = (ua.u + 0x7FFFu + ((ua.u >> 16) & 1u)) >> 16;
    unsigned rb = (ub.u + 0x7FFFu + ((ub.u >> 16) & 1u)) >> 16;
    return (ra & 0xFFFFu) | (rb << 16);
}

// B-fragment: 2x global_load_b128 of bf16 (offsets fold into immediates)
static __device__ __forceinline__ void load_b_frag(FragB16& f, const __bf16* pw, int ks) {
    f.q[0] = *(const uint4*)(pw + ks * 32);
    f.q[1] = *(const uint4*)(pw + ks * 32 + 16);
}

// A-fragment from LDS: lane L holds row M=L%16; kb = ks*32 + 8*(L>=16)
static __device__ __forceinline__ void load_a_frag(FragB16& f, const __bf16* pa, int ks) {
    f.q[0] = *(const uint4*)(pa + ks * 32);
    f.q[1] = *(const uint4*)(pa + ks * 32 + 16);
}

// ---------------------------------------------------------------------------
// Pass 1: convert W to bf16 (coalesced float4 -> uint2), memory bound
// ---------------------------------------------------------------------------
__global__ __launch_bounds__(256)
void convert_w_kernel(const float* __restrict__ W, __bf16* __restrict__ Wb) {
    const int total4 = (WROWS * KDIM) / 4;
    for (int i = blockIdx.x * blockDim.x + threadIdx.x; i < total4;
         i += gridDim.x * blockDim.x) {
        float4 v = ((const float4*)W)[i];
        uint2 p2;
        p2.x = pack2_bf16(v.x, v.y);
        p2.y = pack2_bf16(v.z, v.w);
        *(uint2*)(Wb + (size_t)i * 4) = p2;
    }
}

// ---------------------------------------------------------------------------
// Pass 2: fused GEMM + hypernetwork MLP
// ---------------------------------------------------------------------------
__global__ __launch_bounds__(256)
void fused_hypernet_kernel(const float* __restrict__ int_x,
                           const float* __restrict__ x,
                           const __bf16* __restrict__ Wb,
                           const float* __restrict__ b,
                           float* __restrict__ out) {
    __shared__ __align__(16) __bf16 sA[MTILE * APAD];   // int_x tile bf16 (~33 KB)
    __shared__ __align__(16) float  sH[DDIM * HPAD];    // h^T [p][n]      (~18 KB)

    const int tid    = threadIdx.x;
    const int lane   = tid & 31;
    const int wave   = tid >> 5;          // 0..7
    const int laneLo = lane & 15;
    const int laneHi = lane >> 4;         // 0 or 1
    const int mt     = wave & 1;          // m-tile index (2 x 16 rows)
    const int qt0    = (wave >> 1) * 2;   // first of this wave's 2 q-tiles
    const int n0     = blockIdx.x * MTILE;

    // ---- stage int_x tile: 32x512 f32 -> bf16 LDS (padded rows) ----
    for (int i = tid; i < MTILE * (KDIM / 4); i += 256) {
        int r  = i >> 7;                  // /(KDIM/4)
        int c4 = i & 127;
        float4 v = ((const float4*)(int_x + (size_t)(n0 + r) * KDIM))[c4];
        uint2 p2;
        p2.x = pack2_bf16(v.x, v.y);
        p2.y = pack2_bf16(v.z, v.w);
        *(uint2*)(&sA[r * APAD + c4 * 4]) = p2;
    }
    // ---- stage x tile transposed: h^T[p][n] ----
    for (int e = tid; e < MTILE * DDIM; e += 256) {
        int n = e >> 7;                   // /DDIM
        int p = e & 127;
        sH[p * HPAD + n] = x[(size_t)(n0 + n) * DDIM + p];
    }
    __syncthreads();

    // per-lane bases: A row in LDS; B column offset in a W tile
    const __bf16* paBase = sA + (mt * 16 + laneLo) * APAD + laneHi * 8;
    const size_t  laneOff = (size_t)laneLo * KDIM + laneHi * 8;

    for (int layer = 0; layer < 3; ++layer) {
        const int rowbase = layer * LSTRIDE;

        float acc[2][8];

        // ---- bias tiles (same dual-chain pipelined shape as the p-loop) ----
        {
            const __bf16* pw0 = Wb + (size_t)(rowbase + (qt0    ) * 16) * KDIM + laneOff;
            const __bf16* pw1 = Wb + (size_t)(rowbase + (qt0 + 1) * 16) * KDIM + laneOff;
            v8f c0 = {}, c1 = {};
            FragB16 fa[2], f0[2], f1[2];
            load_a_frag(fa[0], paBase, 0);
            load_b_frag(f0[0], pw0, 0);
            load_b_frag(f1[0], pw1, 0);
#pragma unroll
            for (int ks = 0; ks < KSTEPS; ++ks) {
                const int cur = ks & 1, nxt = cur ^ 1;
                if (ks + 1 < KSTEPS) {
                    load_a_frag(fa[nxt], paBase, ks + 1);
                    load_b_frag(f0[nxt], pw0, ks + 1);
                    load_b_frag(f1[nxt], pw1, ks + 1);
                }
                c0 = __builtin_amdgcn_wmma_f32_16x16x32_bf16(
                         false, fa[cur].v, false, f0[cur].v, (short)0, c0, false, false);
                c1 = __builtin_amdgcn_wmma_f32_16x16x32_bf16(
                         false, fa[cur].v, false, f1[cur].v, (short)0, c1, false, false);
            }
            const float bv0 = b[rowbase + (qt0    ) * 16 + laneLo];
            const float bv1 = b[rowbase + (qt0 + 1) * 16 + laneLo];
#pragma unroll
            for (int j = 0; j < 8; ++j) {
                acc[0][j] = c0[j] + bv0;
                acc[1][j] = c1[j] + bv1;
            }
        }

        // ---- accumulate over the 128 hyper-weight columns p ----
        for (int p = 0; p < DDIM; ++p) {
            if (p + 1 < DDIM)
                __builtin_prefetch(
                    Wb + (size_t)(rowbase + DDIM + (p + 1) * DDIM + qt0 * 16) * KDIM, 0, 1);

            // broadcast h[n, p] for this wave's 16 rows (2 x ds_load_b128)
            const float* ph = &sH[p * HPAD + mt * 16 + laneHi * 8];
            float4 h0 = *(const float4*)(ph);
            float4 h1 = *(const float4*)(ph + 4);
            const float hv[8] = {h0.x, h0.y, h0.z, h0.w, h1.x, h1.y, h1.z, h1.w};

            const int wrow0 = rowbase + DDIM + p * DDIM + (qt0    ) * 16;
            const int wrow1 = rowbase + DDIM + p * DDIM + (qt0 + 1) * 16;
            const __bf16* pw0 = Wb + (size_t)wrow0 * KDIM + laneOff;
            const __bf16* pw1 = Wb + (size_t)wrow1 * KDIM + laneOff;

            v8f c0 = {}, c1 = {};
            FragB16 fa[2], f0[2], f1[2];
            load_a_frag(fa[0], paBase, 0);
            load_b_frag(f0[0], pw0, 0);
            load_b_frag(f1[0], pw1, 0);
#pragma unroll
            for (int ks = 0; ks < KSTEPS; ++ks) {
                const int cur = ks & 1, nxt = cur ^ 1;
                if (ks + 1 < KSTEPS) {
                    load_a_frag(fa[nxt], paBase, ks + 1);
                    load_b_frag(f0[nxt], pw0, ks + 1);
                    load_b_frag(f1[nxt], pw1, ks + 1);
                }
                c0 = __builtin_amdgcn_wmma_f32_16x16x32_bf16(
                         false, fa[cur].v, false, f0[cur].v, (short)0, c0, false, false);
                c1 = __builtin_amdgcn_wmma_f32_16x16x32_bf16(
                         false, fa[cur].v, false, f1[cur].v, (short)0, c1, false, false);
            }
            const float bw0 = b[wrow0 + laneLo];
            const float bw1 = b[wrow1 + laneLo];
#pragma unroll
            for (int j = 0; j < 8; ++j) {
                acc[0][j] += hv[j] * (c0[j] + bw0);
                acc[1][j] += hv[j] * (c1[j] + bw1);
            }
        }

        __syncthreads();   // all reads of current h^T done before overwrite

        if (layer < 2) {
            // ReLU + write h_next back transposed into sH
#pragma unroll
            for (int t = 0; t < 2; ++t) {
                const int q = (qt0 + t) * 16 + laneLo;   // new p index
                float4 lo, hi;
                lo.x = fmaxf(acc[t][0], 0.0f);
                lo.y = fmaxf(acc[t][1], 0.0f);
                lo.z = fmaxf(acc[t][2], 0.0f);
                lo.w = fmaxf(acc[t][3], 0.0f);
                hi.x = fmaxf(acc[t][4], 0.0f);
                hi.y = fmaxf(acc[t][5], 0.0f);
                hi.z = fmaxf(acc[t][6], 0.0f);
                hi.w = fmaxf(acc[t][7], 0.0f);
                float* dst = &sH[q * HPAD + mt * 16 + laneHi * 8];
                *(float4*)(dst)     = lo;
                *(float4*)(dst + 4) = hi;
            }
            __syncthreads();
        } else {
            // final layer: no activation, write to global output
#pragma unroll
            for (int t = 0; t < 2; ++t) {
                const int q = (qt0 + t) * 16 + laneLo;
#pragma unroll
                for (int j = 0; j < 8; ++j) {
                    const int nrow = mt * 16 + laneHi * 8 + j;
                    out[(size_t)(n0 + nrow) * DDIM + q] = acc[t][j];
                }
            }
        }
    }
}

extern "C" void kernel_launch(void* const* d_in, const int* in_sizes, int n_in,
                              void* d_out, int out_size, void* d_ws, size_t ws_size,
                              hipStream_t stream) {
    const float* int_x = (const float*)d_in[0];
    const float* x     = (const float*)d_in[1];
    const float* W     = (const float*)d_in[2];
    const float* b     = (const float*)d_in[3];
    float* out  = (float*)d_out;
    __bf16* Wb  = (__bf16*)d_ws;   // needs WROWS*KDIM*2 = ~50.7 MB of workspace

    // pass 1: W f32 -> bf16 (stream-ordered dependency for pass 2)
    hipLaunchKernelGGL(convert_w_kernel, dim3(2048), dim3(256), 0, stream, W, Wb);

    // pass 2: fused GEMM + MLP
    hipLaunchKernelGGL(fused_hypernet_kernel, dim3(NSAMP / MTILE), dim3(256), 0, stream,
                       int_x, x, Wb, b, out);
}
